// PyGRandLANet_19344532701794
// MI455X (gfx1250) — matrix operand order repaired and checked by reading, not compiled, not run on hardware
//
#include <hip/hip_runtime.h>
#include <hip/hip_bf16.h>

// ============================================================================
// RandLA-Net forward for MI455X (gfx1250, wave32, WMMA).
//  - Dense layers: bf16 WMMA 16x16x32, f32 accumulate. LDS tiles are staged
//    directly in WMMA fragment order so each lane reads its v16bf fragment
//    with two ds_load_b128 (no scalar gather).
//  - KNN: fused distance+top16 in registers; source tiles staged into LDS via
//    GLOBAL_LOAD_ASYNC_TO_LDS_B128 (ASYNCcnt) -- avoids the 268MB distance
//    matrix that would spill the 192MB L2.
//  - Working set ~50MB -> L2 resident; bump allocator over d_ws.
// ============================================================================

#define NEG_SLOPE 0.2f
#define BN_EPS 1e-6f
#define KNN_K 16

typedef __bf16 bf16_t;
typedef bf16_t v16bf __attribute__((ext_vector_type(16)));
typedef float  v8f   __attribute__((ext_vector_type(8)));

__device__ __forceinline__ bf16_t f2bf(float f) {
  union { float f; unsigned u; } v; v.f = f;
  unsigned r = (v.u + 0x7FFFu + ((v.u >> 16) & 1u)) >> 16;   // RNE
  unsigned short s = (unsigned short)r;
  bf16_t b;
  __builtin_memcpy(&b, &s, sizeof(s));
  return b;
}

// ---------------------------------------------------------------------------
// GEMM: Y[M,N] = X[M,K] @ W[K,N] (+ bias). f32 in/out, bf16 WMMA compute.
// 128 threads = 4 waves; workgroup tile 32x32, each wave owns a 16x16 tile.
// LDS layout is fragment-major: [16x16 tile][lane][element].
// ---------------------------------------------------------------------------
__global__ __launch_bounds__(128) void gemm_bias_kernel(
    float* __restrict__ Y, const float* __restrict__ X,
    const float* __restrict__ W, const float* __restrict__ bias,
    int M, int K, int N)
{
  // A fragment mapping (ISA 7.12.2, 16-bit A 16x32): lane = m16 | ((k&8)<<1),
  //   elem = (k&7) | ((k&16)>>1).
  // B fragment mapping (32x16): lane = n16 | ((k&8)<<1), elem = (k&7) | ((k&16)>>1).
  __shared__ __attribute__((aligned(32))) bf16_t Afrag[2][32][16];
  __shared__ __attribute__((aligned(32))) bf16_t Bfrag[2][32][16];

  const int tid  = threadIdx.x;
  const int lane = tid & 31;
  const int wave = tid >> 5;             // 0..3
  const int wm   = (wave >> 1) * 16;     // wave row offset within tile
  const int wn   = (wave & 1) * 16;      // wave col offset within tile
  const int row0 = blockIdx.x * 32;
  const int col0 = blockIdx.y * 32;

  v8f acc = {};

  for (int k0 = 0; k0 < K; k0 += 32) {
    // cooperative stage: 2x1024 elements, 8 per thread each, zero-padded OOB,
    // scattered into fragment order during the f32->bf16 convert.
    for (int i = tid; i < 32 * 32; i += 128) {
      int m = i >> 5, k = i & 31;
      int gm = row0 + m, gk = k0 + k;
      float v = (gm < M && gk < K) ? X[(size_t)gm * K + gk] : 0.0f;
      Afrag[m >> 4][(m & 15) | ((k & 8) << 1)][(k & 7) | ((k & 16) >> 1)] = f2bf(v);
    }
    for (int i = tid; i < 32 * 32; i += 128) {
      int k = i >> 5, n = i & 31;
      int gk = k0 + k, gn = col0 + n;
      float v = (gk < K && gn < N) ? W[(size_t)gk * N + gn] : 0.0f;
      Bfrag[n >> 4][(n & 15) | ((k & 8) << 1)][(k & 7) | ((k & 16) >> 1)] = f2bf(v);
    }
    // prefetch next k-tile while this one computes
    if (k0 + 32 < K) {
      __builtin_prefetch(X + (size_t)(row0 + (tid >> 2)) * K + (k0 + 32), 0, 0);
      __builtin_prefetch(W + (size_t)(k0 + 32 + (tid >> 2)) * N + col0, 0, 0);
    }
    __syncthreads();

    // each lane reads its whole fragment contiguously (2x ds_load_b128 each)
    v16bf afrag = *(const v16bf*)Afrag[wave >> 1][lane];
    v16bf bfrag = *(const v16bf*)Bfrag[wave & 1][lane];

    acc = __builtin_amdgcn_wmma_f32_16x16x32_bf16(
        /*neg_a=*/false, afrag, /*neg_b=*/false, bfrag,
        /*c_mod=*/(short)0, acc, /*reuse_a=*/false, /*reuse_b=*/false);
    __syncthreads();
  }

  // D layout: vgpr r, lane l -> m = r + ((l&16)?8:0), n = l&15
  const int dn   = col0 + wn + (lane & 15);
  const int mofs = (lane & 16) ? 8 : 0;
  const float bb = (bias && dn < N) ? bias[dn] : 0.0f;
#pragma unroll
  for (int r = 0; r < 8; ++r) {
    int dm = row0 + wm + mofs + r;
    if (dm < M && dn < N) Y[(size_t)dm * N + dn] = acc[r] + bb;
  }
}

// ---------------------------------------------------------------------------
// Per-column mean / rsqrt(var+eps). Block covers 32 columns x 8 row-lanes
// so global reads are coalesced across consecutive columns.
// ---------------------------------------------------------------------------
__global__ __launch_bounds__(256) void colstats_kernel(
    float* __restrict__ mean, float* __restrict__ rstd,
    const float* __restrict__ Y, int M, int N)
{
  __shared__ float sh_s[8][32];
  __shared__ float sh_q[8][32];
  const int cl = threadIdx.x & 31;
  const int r  = threadIdx.x >> 5;   // 0..7
  const int c  = blockIdx.x * 32 + cl;
  float s = 0.f, q = 0.f;
  if (c < N) {
    for (int m = r; m < M; m += 8) {
      float v = Y[(size_t)m * N + c];
      s += v; q += v * v;
    }
  }
  sh_s[r][cl] = s; sh_q[r][cl] = q;
  __syncthreads();
  if (r == 0 && c < N) {
#pragma unroll
    for (int j = 1; j < 8; ++j) { s += sh_s[j][cl]; q += sh_q[j][cl]; }
    float mu  = s / (float)M;
    float var = q / (float)M - mu * mu;
    mean[c] = mu;
    rstd[c] = rsqrtf(fmaxf(var, 0.f) + BN_EPS);
  }
}

// y = (y-mean)*rstd*g + bt ; leaky (neg=NEG_SLOPE) or identity (neg=1)
__global__ void bn_act_kernel(float* __restrict__ Y,
                              const float* __restrict__ mean,
                              const float* __restrict__ rstd,
                              const float* __restrict__ g,
                              const float* __restrict__ bt,
                              long long total, int N, float neg)
{
  long long i = (long long)blockIdx.x * 256 + threadIdx.x;
  if (i >= total) return;
  int c = (int)(i % N);
  float v = (Y[i] - mean[c]) * rstd[c] * g[c] + bt[c];
  Y[i] = v >= 0.f ? v : neg * v;
}

// ---------------------------------------------------------------------------
// Fused KNN (top-16 by squared distance). One thread per query; source
// position tile (contiguous 12*nt bytes) staged into LDS with async
// global->LDS DMA (ASYNCcnt), then a sorted 16-entry register list.
// ---------------------------------------------------------------------------
__global__ __launch_bounds__(128) void knn_kernel(
    int* __restrict__ idx_out, const float* __restrict__ qpos,
    const float* __restrict__ spos, int nq, int ns)
{
  __shared__ __attribute__((aligned(16))) float sp[256 * 3];
  const unsigned lds_base = (unsigned)(size_t)&sp[0];   // generic addr low32 = LDS offset

  const int q = blockIdx.x * 128 + threadIdx.x;
  float qx = 0.f, qy = 0.f, qz = 0.f;
  if (q < nq) { qx = qpos[q*3]; qy = qpos[q*3+1]; qz = qpos[q*3+2]; }

  float bd[KNN_K]; int bi[KNN_K];
#pragma unroll
  for (int j = 0; j < KNN_K; ++j) { bd[j] = 3.4e38f; bi[j] = 0; }

  for (int t0 = 0; t0 < ns; t0 += 256) {
    int nt = min(256, ns - t0);
    int nbytes = nt * 12;
    const char* gsrc = (const char*)(spos + (size_t)t0 * 3);

    // async copy: 16B per lane per issue
    for (int off = threadIdx.x * 16; off + 16 <= nbytes; off += 128 * 16) {
      asm volatile("global_load_async_to_lds_b128 %0, %1, off"
                   :: "v"(lds_base + (unsigned)off), "v"(gsrc + off)
                   : "memory");
    }
    // 4B tail (nbytes % 16)
    {
      int tbase = nbytes & ~15;
      int tail4 = (nbytes & 15) >> 2;
      if (threadIdx.x < tail4) {
        int o = tbase + threadIdx.x * 4;
        asm volatile("global_load_async_to_lds_b32 %0, %1, off"
                     :: "v"(lds_base + (unsigned)o), "v"(gsrc + o)
                     : "memory");
      }
    }
    asm volatile("s_wait_asynccnt 0x0" ::: "memory");
    __syncthreads();

    if (q < nq) {
      for (int i = 0; i < nt; ++i) {
        float dx = qx - sp[i*3+0], dy = qy - sp[i*3+1], dz = qz - sp[i*3+2];
        float d  = dx*dx + dy*dy + dz*dz;
        int   ii = t0 + i;
        if (d < bd[KNN_K-1]) {
#pragma unroll
          for (int j = 0; j < KNN_K; ++j) {
            bool sw = d < bd[j];
            float td = bd[j]; int ti = bi[j];
            if (sw) { bd[j] = d; bi[j] = ii; d = td; ii = ti; }
          }
        }
      }
    }
    __syncthreads();
  }
  if (q < nq) {
#pragma unroll
    for (int j = 0; j < KNN_K; ++j) idx_out[q*KNN_K + j] = bi[j];
  }
}

// 1-NN argmin (feature propagation upsampling)
__global__ void nn1_kernel(int* __restrict__ idx_out,
                           const float* __restrict__ qpos,
                           const float* __restrict__ spos, int nq, int ns)
{
  int i = blockIdx.x * 256 + threadIdx.x;
  if (i >= nq) return;
  float qx = qpos[i*3], qy = qpos[i*3+1], qz = qpos[i*3+2];
  float best = 3.4e38f; int bj = 0;
  for (int j = 0; j < ns; ++j) {
    float dx = qx - spos[j*3], dy = qy - spos[j*3+1], dz = qz - spos[j*3+2];
    float d = dx*dx + dy*dy + dz*dz;
    if (d < best) { best = d; bj = j; }
  }
  idx_out[i] = bj;
}

// rel[t] = [pos_i, pos_j, diff, |diff|]  (10 features), t = i*K + j
__global__ void rel_kernel(float* __restrict__ rel, const float* __restrict__ pos,
                           const int* __restrict__ nbr, int n, int k)
{
  int t = blockIdx.x * 256 + threadIdx.x;
  if (t >= n * k) return;
  int i = t / k;
  int j = nbr[t];
  float pix = pos[i*3], piy = pos[i*3+1], piz = pos[i*3+2];
  float pjx = pos[j*3], pjy = pos[j*3+1], pjz = pos[j*3+2];
  float dx = pjx - pix, dy = pjy - piy, dz = pjz - piz;
  float dist = sqrtf(fmaxf(dx*dx + dy*dy + dz*dz, 1e-12f));
  float* o = rel + (size_t)t * 10;
  o[0]=pix; o[1]=piy; o[2]=piz; o[3]=pjx; o[4]=pjy; o[5]=pjz;
  o[6]=dx;  o[7]=dy;  o[8]=dz;  o[9]=dist;
}

// local[(i,j), :] = [ x[nbr[i,j]] (dx cols) , enc[(i,j)] (de cols) ]
__global__ void build_local_kernel(float* __restrict__ local,
                                   const float* __restrict__ x,
                                   const int* __restrict__ nbr,
                                   const float* __restrict__ enc,
                                   long long nk, int dx, int de)
{
  int d = dx + de;
  long long t = (long long)blockIdx.x * 256 + threadIdx.x;
  if (t >= nk * d) return;
  int c = (int)(t % d);
  long long row = t / d;
  float v;
  if (c < dx) v = x[(size_t)nbr[row] * dx + c];
  else        v = enc[row * de + (c - dx)];
  local[t] = v;
}

// softmax over neighbor axis per (point, channel), then weighted sum
__global__ void softmax_agg_kernel(float* __restrict__ out,
                                   const float* __restrict__ att,
                                   const float* __restrict__ local,
                                   int n, int k, int d)
{
  long long t = (long long)blockIdx.x * 256 + threadIdx.x;
  if (t >= (long long)n * d) return;
  int c = (int)(t % d);
  int i = (int)(t / d);
  const float* A = att   + (size_t)i * k * d + c;
  const float* L = local + (size_t)i * k * d + c;
  float mx = -3.4e38f;
  for (int j = 0; j < k; ++j) mx = fmaxf(mx, A[(size_t)j * d]);
  float se = 0.f;
  for (int j = 0; j < k; ++j) se += __expf(A[(size_t)j * d] - mx);
  float inv = 1.f / se, acc = 0.f;
  for (int j = 0; j < k; ++j)
    acc += __expf(A[(size_t)j * d] - mx) * inv * L[(size_t)j * d];
  out[t] = acc;
}

__global__ void add_act_kernel(float* __restrict__ out, const float* __restrict__ h,
                               const float* __restrict__ sc, long long total, float neg)
{
  long long t = (long long)blockIdx.x * 256 + threadIdx.x;
  if (t >= total) return;
  float v = h[t] + sc[t];
  out[t] = v >= 0.f ? v : neg * v;
}

__global__ void concat2_kernel(float* __restrict__ out, const float* __restrict__ a,
                               const float* __restrict__ b, int n, int da, int db)
{
  int d = da + db;
  long long t = (long long)blockIdx.x * 256 + threadIdx.x;
  if (t >= (long long)n * d) return;
  int c = (int)(t % d);
  int i = (int)(t / d);
  out[t] = (c < da) ? a[(size_t)i * da + c] : b[(size_t)i * db + (c - da)];
}

// out[i] = [ src[idx[i]] , skip[i] ]
__global__ void gather_concat_kernel(float* __restrict__ out,
                                     const float* __restrict__ src,
                                     const int* __restrict__ idx,
                                     const float* __restrict__ skip,
                                     int n, int dsrc, int dskip)
{
  int d = dsrc + dskip;
  long long t = (long long)blockIdx.x * 256 + threadIdx.x;
  if (t >= (long long)n * d) return;
  int c = (int)(t % d);
  int i = (int)(t / d);
  out[t] = (c < dsrc) ? src[(size_t)idx[i] * dsrc + c]
                      : skip[(size_t)i * dskip + (c - dsrc)];
}

__global__ void logsoftmax_kernel(float* __restrict__ out,
                                  const float* __restrict__ logits, int n, int c)
{
  int i = blockIdx.x * 256 + threadIdx.x;
  if (i >= n) return;
  const float* L = logits + (size_t)i * c;
  float mx = -3.4e38f;
  for (int j = 0; j < c; ++j) mx = fmaxf(mx, L[j]);
  float se = 0.f;
  for (int j = 0; j < c; ++j) se += __expf(L[j] - mx);
  float ls = mx + __logf(se);
  for (int j = 0; j < c; ++j) out[(size_t)i * c + j] = L[j] - ls;
}

// ============================================================================
// Host-side orchestration
// ============================================================================
namespace {

struct Bump {
  char* base; size_t off; size_t cap;
  void* alloc(size_t bytes) {
    size_t o = (off + 255) & ~(size_t)255;
    off = o + bytes;
    return base + o;
  }
  float* f(size_t n) { return (float*)alloc(n * sizeof(float)); }
  int*   i(size_t n) { return (int*)alloc(n * sizeof(int)); }
};

struct Lay { const float *W, *b, *g, *bt; };
struct Lfa { Lay enc; const float* attW; Lay post; };
struct Blk { Lay mlp1, shortcut, mlp2; Lfa lfa1, lfa2; };

Lay take_lay(void* const* d_in, int& c) {
  Lay l;
  l.W  = (const float*)d_in[c++];
  l.b  = (const float*)d_in[c++];
  l.g  = (const float*)d_in[c++];
  l.bt = (const float*)d_in[c++];
  return l;
}
Lfa take_lfa(void* const* d_in, int& c) {
  Lfa f;
  f.enc  = take_lay(d_in, c);
  f.attW = (const float*)d_in[c++];
  f.post = take_lay(d_in, c);
  return f;
}
Blk take_blk(void* const* d_in, int& c) {
  Blk b;
  b.mlp1     = take_lay(d_in, c);
  b.shortcut = take_lay(d_in, c);
  b.mlp2     = take_lay(d_in, c);
  b.lfa1     = take_lfa(d_in, c);
  b.lfa2     = take_lfa(d_in, c);
  return b;
}

inline unsigned nblk(long long total, int bs) { return (unsigned)((total + bs - 1) / bs); }

void gemm_h(hipStream_t s, float* Y, const float* X, const float* W,
            const float* bias, int M, int K, int N) {
  dim3 g((M + 31) / 32, (N + 31) / 32);
  gemm_bias_kernel<<<g, 128, 0, s>>>(Y, X, W, bias, M, K, N);
}

// y = BN(x@W + b) [+ leaky]. Returns output buffer (Yout if given).
float* apply_layer_h(hipStream_t s, Bump& a, const float* X, int M, int K,
                     const Lay& L, int N, bool act, float* Yout = nullptr) {
  float* Y = Yout ? Yout : a.f((size_t)M * N);
  gemm_h(s, Y, X, L.W, L.b, M, K, N);
  float* mean = a.f(N);
  float* rstd = a.f(N);
  colstats_kernel<<<(N + 31) / 32, 256, 0, s>>>(mean, rstd, Y, M, N);
  long long total = (long long)M * N;
  bn_act_kernel<<<nblk(total, 256), 256, 0, s>>>(Y, mean, rstd, L.g, L.bt,
                                                 total, N, act ? NEG_SLOPE : 1.0f);
  return Y;
}

float* lfa_h(hipStream_t s, Bump& a, const Lfa& P, const int* nbr,
             const float* x, const float* pos, int n, int d) {
  const int k = KNN_K, dx = d / 2, de = d / 2;
  long long nk = (long long)n * k;
  float* rel = a.f(nk * 10);
  rel_kernel<<<nblk(nk, 256), 256, 0, s>>>(rel, pos, nbr, n, k);
  float* enc = apply_layer_h(s, a, rel, (int)nk, 10, P.enc, de, true);
  float* local = a.f(nk * d);
  build_local_kernel<<<nblk(nk * d, 256), 256, 0, s>>>(local, x, nbr, enc, nk, dx, de);
  float* att = a.f(nk * d);
  gemm_h(s, att, local, P.attW, nullptr, (int)nk, d, d);
  float* agg = a.f((size_t)n * d);
  softmax_agg_kernel<<<nblk((long long)n * d, 256), 256, 0, s>>>(agg, att, local, n, k, d);
  return apply_layer_h(s, a, agg, n, d, P.post, d, true);
}

void block_h(hipStream_t s, Bump& a, const Blk& B, const float* x,
             const float* pos, int n, int din, int dout, float* out) {
  int* nbr = a.i((size_t)n * KNN_K);
  knn_kernel<<<(n + 127) / 128, 128, 0, s>>>(nbr, pos, pos, n, n);
  float* sc = apply_layer_h(s, a, x, n, din, B.shortcut, dout, false);
  float* h  = apply_layer_h(s, a, x, n, din, B.mlp1, dout / 8, true);
  h = lfa_h(s, a, B.lfa1, nbr, h, pos, n, dout / 4);
  h = lfa_h(s, a, B.lfa2, nbr, h, pos, n, dout / 2);
  float* h4 = apply_layer_h(s, a, h, n, dout / 2, B.mlp2, dout, false);
  long long total = (long long)n * dout;
  add_act_kernel<<<nblk(total, 256), 256, 0, s>>>(out, h4, sc, total, NEG_SLOPE);
}

void fp_h(hipStream_t s, Bump& a, const Lay& L,
          const float* xsrc, int nsrc, int dsrc, const float* psrc,
          const float* xskip, int nskip, int dskip, const float* pskip,
          int dout, float* out) {
  int* idx = a.i(nskip);
  nn1_kernel<<<nblk(nskip, 256), 256, 0, s>>>(idx, pskip, psrc, nskip, nsrc);
  int dcat = dsrc + dskip;
  float* cat = a.f((size_t)nskip * dcat);
  long long total = (long long)nskip * dcat;
  gather_concat_kernel<<<nblk(total, 256), 256, 0, s>>>(cat, xsrc, idx, xskip,
                                                        nskip, dsrc, dskip);
  apply_layer_h(s, a, cat, nskip, dcat, L, dout, true, out);
}

} // namespace

extern "C" void kernel_launch(void* const* d_in, const int* in_sizes, int n_in,
                              void* d_out, int out_size, void* d_ws, size_t ws_size,
                              hipStream_t stream) {
  (void)in_sizes; (void)n_in; (void)out_size;

  // ---- unpack inputs (setup_inputs() dict insertion order) ----
  const float* pos = (const float*)d_in[0];
  const float* x   = (const float*)d_in[1];
  int c = 2;
  const float* fc0W = (const float*)d_in[c++];
  const float* fc0b = (const float*)d_in[c++];
  Blk b1 = take_blk(d_in, c);
  Blk b2 = take_blk(d_in, c);
  Blk b3 = take_blk(d_in, c);
  Blk b4 = take_blk(d_in, c);
  Lay summit = take_lay(d_in, c);
  Lay fp4L = take_lay(d_in, c);
  Lay fp3L = take_lay(d_in, c);
  Lay fp2L = take_lay(d_in, c);
  Lay fp1L = take_lay(d_in, c);
  Lay end1 = take_lay(d_in, c);
  Lay end2 = take_lay(d_in, c);
  const float* fceW = (const float*)d_in[c++];
  const float* fceb = (const float*)d_in[c++];

  const int n = 8192, n1 = n / 4, n2 = n1 / 4, n3 = n2 / 4, n4 = n3 / 4;

  Bump a{(char*)d_ws, 0, ws_size};

  // ---- persistent buffers (live across stages) ----
  float* cat9 = a.f((size_t)n * 9);
  float* h0   = a.f((size_t)n * 32);
  float* x1   = a.f((size_t)n * 32);     // x1d = first n1 rows (view)
  float* x2   = a.f((size_t)n1 * 128);   // x2d = first n2 rows (view)
  float* x3   = a.f((size_t)n2 * 256);   // x3d = first n3 rows (view)
  float* x4   = a.f((size_t)n3 * 512);   // x4d = first n4 rows (view)
  float* xs   = a.f((size_t)n4 * 512);
  float* f4   = a.f((size_t)n3 * 256);
  float* f3   = a.f((size_t)n2 * 128);
  float* f2   = a.f((size_t)n1 * 32);
  float* f1   = a.f((size_t)n * 32);
  float* e1   = a.f((size_t)n * 64);
  float* e2   = a.f((size_t)n * 32);
  float* lg   = a.f((size_t)n * 7);
  const size_t mark = a.off;

  // ---- stem: h0 = [pos, x] @ fc0 + b (no BN, no act) ----
  concat2_kernel<<<nblk((long long)n * 9, 256), 256, 0, stream>>>(cat9, pos, x, n, 3, 6);
  gemm_h(stream, h0, cat9, fc0W, fc0b, n, 9, 32);

  // ---- encoder blocks (pos prefixes serve as decimated coords) ----
  a.off = mark; block_h(stream, a, b1, h0, pos, n,  32,  32,  x1);
  a.off = mark; block_h(stream, a, b2, x1, pos, n1, 32,  128, x2);
  a.off = mark; block_h(stream, a, b3, x2, pos, n2, 128, 256, x3);
  a.off = mark; block_h(stream, a, b4, x3, pos, n3, 256, 512, x4);

  // ---- summit + feature propagation ----
  a.off = mark;
  apply_layer_h(stream, a, x4, n4, 512, summit, 512, true, xs);
  fp_h(stream, a, fp4L, xs, n4, 512, pos, x3, n3, 256, pos, 256, f4);
  fp_h(stream, a, fp3L, f4, n3, 256, pos, x2, n2, 128, pos, 128, f3);
  fp_h(stream, a, fp2L, f3, n2, 128, pos, x1, n1, 32,  pos, 32,  f2);
  fp_h(stream, a, fp1L, f2, n1, 32,  pos, x1, n,  32,  pos, 32,  f1);

  // ---- head ----
  apply_layer_h(stream, a, f1, n, 32, end1, 64, true, e1);
  apply_layer_h(stream, a, e1, n, 64, end2, 32, true, e2);
  gemm_h(stream, lg, e2, fceW, fceb, n, 32, 7);
  logsoftmax_kernel<<<nblk(n, 256), 256, 0, stream>>>((float*)d_out, lg, n, 7);
}
